// TemporalViT_36850819400472
// MI455X (gfx1250) — compile-verified, matrix operand order checked by reading, not run on hardware
//
#include <hip/hip_runtime.h>
#include <hip/hip_bf16.h>
#include <math.h>

typedef _Float16 h8  __attribute__((ext_vector_type(8)));
typedef _Float16 h16 __attribute__((ext_vector_type(16)));
typedef float    f8  __attribute__((ext_vector_type(8)));

namespace {
constexpr int cB = 4, cT = 32, cD = 768, cH = 12, cL = 3, cFF = 3072, cN = 37;
constexpr int cFrames = cB * cT;      // 128
constexpr int cNP = 36;
constexpr int cTOK = cFrames * cN;    // 4736 tokens (same count in both stages)
constexpr int cKP = 608;              // patch K 588 padded to 19*32
constexpr long long cSMAX = 48LL * 1184 * 1184;   // largest attention score matrix
}

// CDNA5 async DMA: copy 16 contiguous bytes global -> LDS, tracked by ASYNCcnt.
__device__ __forceinline__ void async_cp16(void* lds, const void* g) {
  unsigned l = (unsigned)(unsigned long long)(uintptr_t)lds;   // LDS offset = addr[31:0]
  unsigned long long ga = (unsigned long long)(uintptr_t)g;
  asm volatile("global_load_async_to_lds_b128 %0, %1, off" :: "v"(l), "v"(ga) : "memory");
}
__device__ __forceinline__ void wait_async0() {
  asm volatile("s_wait_asynccnt 0x0" ::: "memory");
}

// ---------------------------------------------------------------------------
// Tiled WMMA GEMM. 256 threads = 8 wave32 waves (WR x WC); BK = 32.
// Fragments load as two contiguous ds_load_b128 per the CDNA5 16-bit layout:
//   lanes 0-15: K 0..7 (vgpr0-3) and 16..23 (vgpr4-7); lanes 16-31: +8.
//
// GUARD=false (all big GEMMs): B is PRE-TRANSPOSED [N][K]; M%BM==0, N%BN==0,
//   K%32==0, rows 16B aligned. Staging = global_load_async_to_lds_b128 with
//   double-buffered LDS (DMA of tile k+1 overlaps WMMA of tile k); staging
//   loops have compile-time trip counts -> no exec masking; last K-iteration
//   peeled -> branch-free steady state.
// GUARD=true (attention): B is [K][N]; register staging with bounds checks +
//   in-LDS transpose of B; handles any M/N/K and unaligned rows.
// ---------------------------------------------------------------------------
template<int BM, int BN, int WR, int WC, bool FGELU, bool GUARD>
__global__ __launch_bounds__(256) void k_gemm(
    const _Float16* __restrict__ A, long long sA,
    const _Float16* __restrict__ Bm, long long sB,
    float* C, long long sC,
    _Float16* Ch, long long sCh,
    const float* __restrict__ bias,
    const float* res, long long sRes,
    int M, int N, int K)
{
  constexpr int TM = BM / WR, TN = BN / WC, FM = TM / 16, FN = TN / 16;
  constexpr int NBUF = GUARD ? 1 : 2;
  __shared__ _Float16 As[NBUF][BM][32];
  __shared__ _Float16 Bs[NBUF][BN][32];

  const int tid  = threadIdx.x;
  const int wave = tid >> 5, lane = tid & 31;
  const int wr = wave / WC, wc = wave % WC;
  const int l15 = lane & 15;
  const int hi8 = (lane >> 4) * 8;          // K (and C-row) offset for upper half-wave
  const int mBase = blockIdx.y * BM, nBase = blockIdx.x * BN;
  const long long bz = blockIdx.z;
  A  += bz * sA;
  Bm += bz * sB;
  if (C)   C   += bz * sC;
  if (Ch)  Ch  += bz * sCh;
  if (res) res += bz * sRes;

  f8 acc[FM][FN];
#pragma unroll
  for (int i = 0; i < FM; i++)
#pragma unroll
    for (int j = 0; j < FN; j++)
#pragma unroll
      for (int e = 0; e < 8; e++) acc[i][j][e] = 0.f;

  auto compute = [&](int b) {
    h16 af[FM], bf[FN];
#pragma unroll
    for (int fm = 0; fm < FM; fm++) {
      const _Float16* pA = &As[b][wr * TM + fm * 16 + l15][hi8];
      h8 lo = *(const h8*)pA;
      h8 hv = *(const h8*)(pA + 16);
      af[fm] = __builtin_shufflevector(lo, hv, 0,1,2,3,4,5,6,7,8,9,10,11,12,13,14,15);
    }
#pragma unroll
    for (int fn = 0; fn < FN; fn++) {
      const _Float16* pB = &Bs[b][wc * TN + fn * 16 + l15][hi8];
      h8 lo = *(const h8*)pB;
      h8 hv = *(const h8*)(pB + 16);
      bf[fn] = __builtin_shufflevector(lo, hv, 0,1,2,3,4,5,6,7,8,9,10,11,12,13,14,15);
    }
#pragma unroll
    for (int fm = 0; fm < FM; fm++)
#pragma unroll
      for (int fn = 0; fn < FN; fn++)
        acc[fm][fn] = __builtin_amdgcn_wmma_f32_16x16x32_f16(
            false, af[fm], false, bf[fn], (short)0, acc[fm][fn], false, false);
  };

  if constexpr (!GUARD) {
    // Both A ([M][K]) and Bt ([N][K]) tiles are contiguous 16B chunks -> async DMA.
    // Per-thread chunk coordinates are compile-time-shaped: no bound tests.
    const int ar = tid >> 2, ak = (tid & 3) << 3;   // + j*64 rows per extra chunk
    auto issue = [&](int k0, int b) {
      constexpr int CA = (BM * 4) / 256;
#pragma unroll
      for (int j = 0; j < CA; j++) {
        int r = ar + j * 64;
        async_cp16(&As[b][r][ak], A + (long long)(mBase + r) * K + (k0 + ak));
      }
      constexpr int CBc = (BN * 4) / 256;
#pragma unroll
      for (int j = 0; j < CBc; j++) {
        int r = ar + j * 64;
        async_cp16(&Bs[b][r][ak], Bm + (long long)(nBase + r) * K + (k0 + ak));
      }
    };
    const int nk = K >> 5;                 // K % 32 == 0
    issue(0, 0);
    for (int i = 0; i < nk - 1; i++) {     // branch-free steady state
      wait_async0();
      __syncthreads();                     // tile i fully in LDS for all waves
      issue((i + 1) * 32, (i + 1) & 1);    // DMA next tile while computing
      compute(i & 1);
      __syncthreads();                     // all waves done reading buffer i&1
    }
    wait_async0();
    __syncthreads();
    compute((nk - 1) & 1);                 // peeled tail
  } else {
    for (int k0 = 0; k0 < K; k0 += 32) {
      // ---- stage A tile: BM x 32, 8-half chunks, zero-padded ----
#pragma unroll
      for (int c = tid; c < BM * 4; c += 256) {
        int r = c >> 2, kc = (c & 3) << 3;
        int gm = mBase + r, gk = k0 + kc;
        h8 v;
#pragma unroll
        for (int i = 0; i < 8; i++) v[i] = (_Float16)0;
        if (gm < M && gk < K) {
          const _Float16* src = A + (long long)gm * K + gk;
          if (gk + 8 <= K && (((unsigned long long)src & 15ull) == 0)) {
            v = *(const h8*)src;
          } else {
            for (int i = 0; i < 8; i++) if (gk + i < K) v[i] = src[i];
          }
        }
        *(h8*)&As[0][r][kc] = v;
      }
      // ---- stage B tile transposed into Bs[n][k] ----
#pragma unroll
      for (int c = tid; c < BN * 4; c += 256) {
        int k = c & 31, n0 = (c >> 5) << 3;
        int gk = k0 + k, gn = nBase + n0;
        h8 v;
#pragma unroll
        for (int i = 0; i < 8; i++) v[i] = (_Float16)0;
        if (gk < K && gn < N) {
          const _Float16* src = Bm + (long long)gk * N + gn;
          if (gn + 8 <= N && (((unsigned long long)src & 15ull) == 0)) {
            v = *(const h8*)src;
          } else {
            for (int i = 0; i < 8; i++) if (gn + i < N) v[i] = src[i];
          }
        }
#pragma unroll
        for (int i = 0; i < 8; i++) Bs[0][n0 + i][k] = v[i];
      }
      __syncthreads();
      compute(0);
      __syncthreads();
    }
  }

  // ---- epilogue: C layout = lanes 0-15 N=lane M=v, lanes 16-31 N=lane-16 M=8+v ----
  const int mW = mBase + wr * TM, nW = nBase + wc * TN;
#pragma unroll
  for (int fm = 0; fm < FM; fm++) {
#pragma unroll
    for (int fn = 0; fn < FN; fn++) {
      int col = nW + fn * 16 + l15;
      if (GUARD && col >= N) continue;
      float bv = bias ? bias[col] : 0.f;
      int rb = mW + fm * 16 + hi8;
#pragma unroll
      for (int v = 0; v < 8; v++) {
        int row = rb + v;
        if (GUARD && row >= M) continue;
        float x = acc[fm][fn][v] + bv;
        if (FGELU) x = 0.5f * x * (1.f + erff(x * 0.70710678118654752f));
        long long o = (long long)row * N + col;
        if (res) x += res[o];
        if (C)  C[o]  = x;
        if (Ch) Ch[o] = (_Float16)x;
      }
    }
  }
}

// ---------------------------------------------------------------------------
// Elementwise / reduction helpers
// ---------------------------------------------------------------------------
// Cast f32 [K][N] weight -> f16 TRANSPOSED [N][K] (for the async GEMM path).
__global__ void k_cast_t_f16(const float* __restrict__ in, _Float16* __restrict__ out,
                             int K, int N) {
  long long i = (long long)blockIdx.x * 256 + threadIdx.x;
  if (i >= (long long)K * N) return;
  int n = (int)(i / K), k = (int)(i % K);
  out[i] = (_Float16)in[(long long)k * N + n];
}

// conv_w (768, 588) is naturally [N][K]; pad K 588 -> 608 with zeros.
__global__ void k_convw_t(const float* __restrict__ w, _Float16* __restrict__ Bt) {
  int i = blockIdx.x * 256 + threadIdx.x;
  if (i >= cD * cKP) return;
  int n = i / cKP, k = i % cKP;
  Bt[i] = (k < 588) ? (_Float16)w[(long long)n * 588 + k] : (_Float16)0;
}

__global__ void k_im2col(const float* __restrict__ x, _Float16* __restrict__ A) {
  long long i = (long long)blockIdx.x * 256 + threadIdx.x;
  if (i >= (long long)cFrames * cNP * cKP) return;
  int k = (int)(i % cKP);
  int row = (int)(i / cKP);
  int frame = row / cNP, patch = row % cNP;
  _Float16 v = (_Float16)0;
  if (k < 588) {
    int c = k / 196, rem = k % 196, iy = rem / 14, ix = rem % 14;
    int py = patch / 6, px = patch % 6;
    v = (_Float16)x[(((long long)frame * 3 + c) * 84 + (py * 14 + iy)) * 84 + (px * 14 + ix)];
  }
  A[i] = v;
}

__global__ void k_assemble(const float* __restrict__ PC, const float* __restrict__ cls,
                           const float* __restrict__ pos, float* __restrict__ X) {
  long long i = (long long)blockIdx.x * 256 + threadIdx.x;
  if (i >= (long long)cTOK * cD) return;
  int c = (int)(i % cD);
  int tok = (int)(i / cD);
  int frame = tok / cN, n = tok % cN;
  float v;
  if (n == 0) v = cls[c] + pos[c];
  else        v = PC[((long long)frame * cNP + (n - 1)) * cD + c] + pos[(long long)n * cD + c];
  X[i] = v;
}

// LayerNorm over D=768; one block (256 thr) per row; writes f32 and/or f16.
__global__ __launch_bounds__(256) void k_layernorm(const float* __restrict__ X,
    const float* __restrict__ g, const float* __restrict__ b,
    float* __restrict__ Yf, _Float16* __restrict__ Yh)
{
  __shared__ float red[256];
  long long row = blockIdx.x;
  const float* x = X + row * cD;
  int t = threadIdx.x;
  float v0 = x[t], v1 = x[t + 256], v2 = x[t + 512];
  red[t] = v0 + v1 + v2; __syncthreads();
  for (int o = 128; o > 0; o >>= 1) { if (t < o) red[t] += red[t + o]; __syncthreads(); }
  float mean = red[0] * (1.f / cD); __syncthreads();
  red[t] = v0 * v0 + v1 * v1 + v2 * v2; __syncthreads();
  for (int o = 128; o > 0; o >>= 1) { if (t < o) red[t] += red[t + o]; __syncthreads(); }
  float var = red[0] * (1.f / cD) - mean * mean;
  float rstd = rsqrtf(var + 1e-5f);
#pragma unroll
  for (int j = 0; j < 3; j++) {
    int c = t + j * 256;
    float y = (x[c] - mean) * rstd * g[c] + b[c];
    if (Yf) Yf[row * cD + c] = y;
    if (Yh) Yh[row * cD + c] = (_Float16)y;
  }
}

__global__ void k_split_qkv(const float* __restrict__ qkv, _Float16* __restrict__ Q,
                            _Float16* __restrict__ Kt, _Float16* __restrict__ V, int SEQ) {
  long long i = (long long)blockIdx.x * 256 + threadIdx.x;
  if (i >= (long long)cTOK * cD) return;
  int c = (int)(i % cD), tok = (int)(i / cD);
  int hh = c >> 6, d = c & 63;
  int sb = tok / SEQ, n = tok % SEQ;
  long long bh = (long long)sb * cH + hh;
  const float* src = qkv + (long long)tok * (3 * cD);
  Q [(bh * SEQ + n) * 64 + d]  = (_Float16)(src[c] * 0.125f);   // HD^-0.5 = 1/8
  Kt[(bh * 64 + d) * SEQ + n]  = (_Float16)src[cD + c];
  V [(bh * SEQ + n) * 64 + d]  = (_Float16)src[2 * cD + c];
}

__global__ __launch_bounds__(256) void k_softmax(const float* __restrict__ S,
                                                 _Float16* __restrict__ P, int len) {
  __shared__ float red[256];
  long long row = blockIdx.x;
  const float* s = S + row * (long long)len;
  _Float16* p = P + row * (long long)len;
  int t = threadIdx.x;
  float m = -3.4e38f;
  for (int c = t; c < len; c += 256) m = fmaxf(m, s[c]);
  red[t] = m; __syncthreads();
  for (int o = 128; o > 0; o >>= 1) { if (t < o) red[t] = fmaxf(red[t], red[t + o]); __syncthreads(); }
  m = red[0]; __syncthreads();
  float sum = 0.f;
  for (int c = t; c < len; c += 256) sum += expf(s[c] - m);
  red[t] = sum; __syncthreads();
  for (int o = 128; o > 0; o >>= 1) { if (t < o) red[t] += red[t + o]; __syncthreads(); }
  float inv = 1.f / red[0];
  for (int c = t; c < len; c += 256) p[c] = (_Float16)(expf(s[c] - m) * inv);
}

__global__ void k_merge_heads(const float* __restrict__ O, _Float16* __restrict__ out, int SEQ) {
  long long i = (long long)blockIdx.x * 256 + threadIdx.x;
  if (i >= (long long)cTOK * cD) return;
  int c = (int)(i % cD), tok = (int)(i / cD);
  int hh = c >> 6, d = c & 63;
  int sb = tok / SEQ, n = tok % SEQ;
  long long bh = (long long)sb * cH + hh;
  out[i] = (_Float16)O[(bh * SEQ + n) * 64 + d];
}

__global__ void k_add_tpos(float* X, const float* __restrict__ tpos) {
  long long i = (long long)blockIdx.x * 256 + threadIdx.x;
  if (i >= (long long)cTOK * cD) return;
  int c = (int)(i % cD), tok = (int)(i / cD);
  int r = tok % (cT * cN);
  int t = r / cN;
  X[i] += tpos[(long long)t * cD + c];
}

__global__ void k_add(float* x, const float* __restrict__ y, long long n) {
  long long i = (long long)blockIdx.x * 256 + threadIdx.x;
  if (i < n) x[i] += y[i];
}

__global__ void k_mean(const float* __restrict__ X, float* __restrict__ out) {
  int i = blockIdx.x * 256 + threadIdx.x;
  if (i >= cB * cD) return;
  int b = i / cD, c = i % cD;
  const float* p = X + (long long)b * (cT * cN) * cD + c;
  float s = 0.f;
  for (int r = 0; r < cT * cN; r++) s += p[(long long)r * cD];
  out[i] = s * (1.0f / (cT * cN));
}

// ---------------------------------------------------------------------------
// Host-side launch helpers
// ---------------------------------------------------------------------------
static inline unsigned egrid(long long n) { return (unsigned)((n + 255) / 256); }

// Big GEMMs: exactly tile-divisible, B pre-transposed [N][K] -> async path.
static void gemm_big(hipStream_t st, const _Float16* A, const _Float16* Bt,
                     float* C, _Float16* Ch, const float* bias, const float* res,
                     int M, int N, int K, bool gelu) {
  dim3 g((N + 127) / 128, (M + 127) / 128, 1), blk(256);
  if (gelu)
    k_gemm<128,128,4,2,true ,false><<<g, blk, 0, st>>>(A, 0, Bt, 0, C, 0, Ch, 0, bias, res, 0, M, N, K);
  else
    k_gemm<128,128,4,2,false,false><<<g, blk, 0, st>>>(A, 0, Bt, 0, C, 0, Ch, 0, bias, res, 0, M, N, K);
}

static void gemm_attn(hipStream_t st, const _Float16* A, long long sA,
                      const _Float16* B, long long sB, float* C, long long sC,
                      int M, int N, int K, int batch) {
  dim3 g((N + 63) / 64, (M + 63) / 64, batch), blk(256);
  k_gemm<64,64,2,4,false,true><<<g, blk, 0, st>>>(A, sA, B, sB, C, sC,
                                                  (_Float16*)nullptr, 0, nullptr, nullptr, 0, M, N, K);
}

struct LW {
  const float *pre_g, *pre_b, *n1_g, *n1_b, *qkv_w, *qkv_b, *proj_w, *proj_b,
              *n2_g, *n2_b, *fc1_w, *fc1_b, *fc2_w, *fc2_b;
};

static void run_block(hipStream_t st, const LW& w, int SEQ,
                      float* X, float* XN, float* Y, float* QKV, float* OH,
                      _Float16* BUFH, _Float16* HH, _Float16* WH,
                      _Float16* QH, _Float16* KTH, _Float16* VH,
                      float* S, _Float16* PH) {
  const int NSEQ = cTOK / SEQ;
  const int BH = NSEQ * cH;
  const long long nTD = (long long)cTOK * cD;

  // xn = LN(x, pre)
  k_layernorm<<<cTOK, 256, 0, st>>>(X, w.pre_g, w.pre_b, XN, nullptr);
  // ln1(xn) -> f16
  k_layernorm<<<cTOK, 256, 0, st>>>(XN, w.n1_g, w.n1_b, nullptr, BUFH);
  // qkv GEMM (weight cast+transpose -> [2304][768])
  k_cast_t_f16<<<egrid((long long)cD * 3 * cD), 256, 0, st>>>(w.qkv_w, WH, cD, 3 * cD);
  gemm_big(st, BUFH, WH, QKV, nullptr, w.qkv_b, nullptr, cTOK, 3 * cD, cD, false);
  // split into per-head Q (scaled), K^T, V
  k_split_qkv<<<egrid(nTD), 256, 0, st>>>(QKV, QH, KTH, VH, SEQ);
  // scores = Q @ K^T  (batched over heads)
  gemm_attn(st, QH, (long long)SEQ * 64, KTH, (long long)SEQ * 64,
            S, (long long)SEQ * SEQ, SEQ, SEQ, 64, BH);
  // softmax rows -> f16 probs
  k_softmax<<<(unsigned)((long long)BH * SEQ), 256, 0, st>>>(S, PH, SEQ);
  // out = P @ V
  gemm_attn(st, PH, (long long)SEQ * SEQ, VH, (long long)SEQ * 64,
            OH, (long long)SEQ * 64, SEQ, 64, SEQ, BH);
  // merge heads -> f16
  k_merge_heads<<<egrid(nTD), 256, 0, st>>>(OH, BUFH, SEQ);
  // y = xn + proj(attn)   (weight -> [768][768] transposed)
  k_cast_t_f16<<<egrid((long long)cD * cD), 256, 0, st>>>(w.proj_w, WH, cD, cD);
  gemm_big(st, BUFH, WH, Y, nullptr, w.proj_b, XN, cTOK, cD, cD, false);
  // h = GELU(LN(y,n2) @ fc1 + b)  (f16 out only; weight -> [3072][768])
  k_layernorm<<<cTOK, 256, 0, st>>>(Y, w.n2_g, w.n2_b, nullptr, BUFH);
  k_cast_t_f16<<<egrid((long long)cD * cFF), 256, 0, st>>>(w.fc1_w, WH, cD, cFF);
  gemm_big(st, BUFH, WH, nullptr, HH, w.fc1_b, nullptr, cTOK, cFF, cD, true);
  // y = y + h @ fc2 + b   (residual fused, in-place on Y; weight -> [768][3072])
  k_cast_t_f16<<<egrid((long long)cFF * cD), 256, 0, st>>>(w.fc2_w, WH, cFF, cD);
  gemm_big(st, HH, WH, Y, nullptr, w.fc2_b, Y, cTOK, cD, cFF, false);
  // x = x + y
  k_add<<<egrid(nTD), 256, 0, st>>>(X, Y, nTD);
}

static inline void* wsa(char*& p, long long bytes) {
  void* r = p; p += (bytes + 255) & ~255LL; return r;
}

extern "C" void kernel_launch(void* const* d_in, const int* in_sizes, int n_in,
                              void* d_out, int out_size, void* d_ws, size_t ws_size,
                              hipStream_t stream) {
  (void)in_sizes; (void)n_in; (void)out_size; (void)ws_size;
  const float* x      = (const float*)d_in[0];
  const float* conv_w = (const float*)d_in[1];
  const float* conv_b = (const float*)d_in[2];
  const float* cls    = (const float*)d_in[3];
  const float* pos    = (const float*)d_in[4];
  const float* tpos   = (const float*)d_in[5];

  LW sw[cL], tw[cL];
  for (int stg = 0; stg < 2; stg++) {
    const float* const* q = (const float* const*)(d_in + 6 + stg * 14);
    for (int i = 0; i < cL; i++) {
      LW& w = (stg == 0 ? sw[i] : tw[i]);
      w.pre_g  = q[0]  + (long long)i * cD;
      w.pre_b  = q[1]  + (long long)i * cD;
      w.n1_g   = q[2]  + (long long)i * cD;
      w.n1_b   = q[3]  + (long long)i * cD;
      w.qkv_w  = q[4]  + (long long)i * cD * 3 * cD;
      w.qkv_b  = q[5]  + (long long)i * 3 * cD;
      w.proj_w = q[6]  + (long long)i * cD * cD;
      w.proj_b = q[7]  + (long long)i * cD;
      w.n2_g   = q[8]  + (long long)i * cD;
      w.n2_b   = q[9]  + (long long)i * cD;
      w.fc1_w  = q[10] + (long long)i * cD * cFF;
      w.fc1_b  = q[11] + (long long)i * cFF;
      w.fc2_w  = q[12] + (long long)i * cFF * cD;
      w.fc2_b  = q[13] + (long long)i * cD;
    }
  }

  // ---- workspace carve-out ----
  char* p = (char*)d_ws;
  float*     X    = (float*)    wsa(p, (long long)cTOK * cD * 4);
  float*     XN   = (float*)    wsa(p, (long long)cTOK * cD * 4);
  float*     Y    = (float*)    wsa(p, (long long)cTOK * cD * 4);
  float*     QKV  = (float*)    wsa(p, (long long)cTOK * 3 * cD * 4);   // also conv out
  float*     OH   = (float*)    wsa(p, (long long)cTOK * cD * 4);
  _Float16*  BUFH = (_Float16*) wsa(p, (long long)cTOK * cD * 2);
  _Float16*  HH   = (_Float16*) wsa(p, (long long)cTOK * cFF * 2);      // also im2col A
  _Float16*  WH   = (_Float16*) wsa(p, (long long)cD * cFF * 2);
  _Float16*  QH   = (_Float16*) wsa(p, (long long)cTOK * cD * 2);
  _Float16*  KTH  = (_Float16*) wsa(p, (long long)cTOK * cD * 2);
  _Float16*  VH   = (_Float16*) wsa(p, (long long)cTOK * cD * 2);
  float*     S    = (float*)    wsa(p, cSMAX * 4);
  _Float16*  PH   = (_Float16*) wsa(p, cSMAX * 2);

  // ---- patch embedding: im2col + WMMA GEMM (4608 x 608 x 768) ----
  k_convw_t<<<egrid((long long)cD * cKP), 256, 0, stream>>>(conv_w, WH);
  k_im2col<<<egrid((long long)cFrames * cNP * cKP), 256, 0, stream>>>(x, HH);
  gemm_big(stream, HH, WH, QKV /*conv out 4608x768*/, nullptr, conv_b, nullptr,
           cFrames * cNP, cD, cKP, false);
  k_assemble<<<egrid((long long)cTOK * cD), 256, 0, stream>>>(QKV, cls, pos, X);

  // ---- spatial transformer: 128 sequences of 37 tokens ----
  for (int i = 0; i < cL; i++)
    run_block(stream, sw[i], cN, X, XN, Y, QKV, OH, BUFH, HH, WH, QH, KTH, VH, S, PH);

  // ---- reshape (no-op in memory) + temporal positions ----
  k_add_tpos<<<egrid((long long)cTOK * cD), 256, 0, stream>>>(X, tpos);

  // ---- temporal transformer: 4 sequences of 1184 tokens ----
  for (int i = 0; i < cL; i++)
    run_block(stream, tw[i], cT * cN, X, XN, Y, QKV, OH, BUFH, HH, WH, QH, KTH, VH, S, PH);

  // ---- mean over tokens -> (4, 768) ----
  k_mean<<<egrid(cB * cD), 256, 0, stream>>>(X, (float*)d_out);
}
